// Loss_2482491097912
// MI455X (gfx1250) — matrix operand order repaired and checked by reading, not compiled, not run on hardware
//
#include <hip/hip_runtime.h>

// MI455X / gfx1250, wave32, WMMA f16 16x16x32 with f32 accumulation.

typedef __attribute__((ext_vector_type(16))) _Float16 v16h;
typedef __attribute__((ext_vector_type(8)))  _Float16 v8h;
typedef __attribute__((ext_vector_type(8)))  float    v8f;

#define N_ROWS 8192
#define D_DIM  512

__device__ __forceinline__ v16h cat8(v8h lo, v8h hi) {
  return __builtin_shufflevector(lo, hi, 0,1,2,3,4,5,6,7,8,9,10,11,12,13,14,15);
}

// Wave computes a 32x64 tile of X (MxK) @ Y^T (NxK), K multiple of 32.
// acc[mf][nf] is the 16x16 f32 accumulator for rows rowBase+mf*16.., cols colBase+nf*16..
__device__ __forceinline__ void wave_mma_nt(const _Float16* __restrict__ X,
                                            const _Float16* __restrict__ Y,
                                            int rowBase, int colBase, int K,
                                            int lane, v8f acc[2][4]) {
  const int lhi = lane >> 4;     // 0 or 1 (which half of the wave)
  const int llo = lane & 15;
  const _Float16* xa0 = X + (size_t)(rowBase +  0 + llo) * K;
  const _Float16* xa1 = X + (size_t)(rowBase + 16 + llo) * K;
  const _Float16* yb  = Y + (size_t)(colBase + llo) * K;
#pragma unroll 4
  for (int k0 = 0; k0 < K; k0 += 32) {
    v16h a[2], b[4];
    {
      // A fragment (16x32 f16): lane L holds row (L&15);
      // halves [0..7] = K (L>>4)*8+0..7, halves [8..15] = K 16+(L>>4)*8+0..7
      const _Float16* p = xa0 + k0 + lhi * 8;
      a[0] = cat8(*(const v8h*)p, *(const v8h*)(p + 16));
      p = xa1 + k0 + lhi * 8;
      a[1] = cat8(*(const v8h*)p, *(const v8h*)(p + 16));
    }
#pragma unroll
    for (int nf = 0; nf < 4; ++nf) {
      // B fragment (32x16 f16), B[k,n] = Y[colBase+n, k]:
      // lane L holds col (L&15); halves [0..7] = K (L>>4)*8+0..7, halves [8..15] = K 16+(L>>4)*8+0..7
      const _Float16* p = yb + (size_t)(nf * 16) * K + k0 + lhi * 8;
      b[nf] = cat8(*(const v8h*)p, *(const v8h*)(p + 16));
    }
#pragma unroll
    for (int mf = 0; mf < 2; ++mf)
#pragma unroll
      for (int nf = 0; nf < 4; ++nf)
        acc[mf][nf] = __builtin_amdgcn_wmma_f32_16x16x32_f16(
            false, a[mf], false, b[nf], (short)0, acc[mf][nf], false, false);
  }
}

__global__ void f32_to_h(const float* __restrict__ in, _Float16* __restrict__ out, int n) {
  int i = blockIdx.x * blockDim.x + threadIdx.x;
  if (i < n) out[i] = (_Float16)in[i];
}

// MODE 0: out = elu(c + bias) as f16 ; MODE 1: out = c + bias as f32
template <int MODE>
__global__ __launch_bounds__(256) void gemm_nt_bias(const _Float16* __restrict__ X,
                                                    const _Float16* __restrict__ W,
                                                    const float* __restrict__ bias,
                                                    _Float16* __restrict__ outH,
                                                    float* __restrict__ outF,
                                                    int M, int NC, int K) {
  const int lane = threadIdx.x & 31;
  const int wave = threadIdx.x >> 5;
  const int waveRow = wave >> 1, waveCol = wave & 1;
  const int rowBase = blockIdx.y * 128 + waveRow * 32;
  const int colBase = blockIdx.x * 128 + waveCol * 64;
  v8f acc[2][4] = {};
  wave_mma_nt(X, W, rowBase, colBase, K, lane, acc);
  const int lhi = lane >> 4, llo = lane & 15;
#pragma unroll
  for (int mf = 0; mf < 2; ++mf)
#pragma unroll
    for (int nf = 0; nf < 4; ++nf) {
      v8f a = acc[mf][nf];
#pragma unroll
      for (int v = 0; v < 8; ++v) {
        // C/D layout: VGPR v, lanes 0-15 -> M=v, lanes 16-31 -> M=v+8; N=lane&15
        int grow = rowBase + mf * 16 + v + 8 * lhi;
        int gcol = colBase + nf * 16 + llo;
        float c = a[v] + bias[gcol];
        if (MODE == 0) {
          float o = c > 0.f ? c : expm1f(c);
          outH[(size_t)grow * NC + gcol] = (_Float16)o;
        } else {
          outF[(size_t)grow * NC + gcol] = c;
        }
      }
    }
}

__global__ __launch_bounds__(256) void normalize_rows(const float* __restrict__ H,
                                                      _Float16* __restrict__ out, int D) {
  __shared__ float red[256];
  const int row = blockIdx.x;
  const float* h = H + (size_t)row * D;
  float ss = 0.f;
  for (int j = threadIdx.x; j < D; j += 256) { float x = h[j]; ss += x * x; }
  red[threadIdx.x] = ss;
  __syncthreads();
  for (int s = 128; s > 0; s >>= 1) {
    if (threadIdx.x < s) red[threadIdx.x] += red[threadIdx.x + s];
    __syncthreads();
  }
  float inv = 1.0f / fmaxf(sqrtf(red[0]), 1e-12f);
  for (int j = threadIdx.x; j < D; j += 256)
    out[(size_t)row * D + j] = (_Float16)(h[j] * inv);
}

// One 128x128 tile of exp(2 * X@Y^T); fused reductions:
//  - per-row partial sums -> partialRow[colTile*N + row]
//  - diagonal values -> diagOut (if non-null)
//  - mask-weighted sums (both orientations) -> maskPartial[tile*4..] (if posM non-null)
__global__ __launch_bounds__(256) void sim_pass(const _Float16* __restrict__ X,
                                                const _Float16* __restrict__ Y,
                                                float* __restrict__ partialRow,
                                                float* __restrict__ diagOut,
                                                const unsigned char* __restrict__ posM,
                                                const unsigned char* __restrict__ negM,
                                                float* __restrict__ maskPartial,
                                                int N, int K) {
  __shared__ float rowAcc[128];
  __shared__ float mAcc[4];
  if (threadIdx.x < 128) rowAcc[threadIdx.x] = 0.f;
  if (threadIdx.x < 4) mAcc[threadIdx.x] = 0.f;
  __syncthreads();

  const int lane = threadIdx.x & 31;
  const int wave = threadIdx.x >> 5;
  const int waveRow = wave >> 1, waveCol = wave & 1;
  const int tileRow = blockIdx.y * 128, tileCol = blockIdx.x * 128;
  const int rowBase = tileRow + waveRow * 32;
  const int colBase = tileCol + waveCol * 64;

  v8f acc[2][4] = {};
  wave_mma_nt(X, Y, rowBase, colBase, K, lane, acc);

  const int lhi = lane >> 4, llo = lane & 15;
  float pP = 0.f, pN = 0.f, pPT = 0.f, pNT = 0.f;
#pragma unroll
  for (int mf = 0; mf < 2; ++mf)
#pragma unroll
    for (int nf = 0; nf < 4; ++nf) {
      v8f a = acc[mf][nf];
#pragma unroll
      for (int v = 0; v < 8; ++v) {
        int rl = waveRow * 32 + mf * 16 + v + 8 * lhi;   // row within 128-tile
        int grow = tileRow + rl;
        int gcol = colBase + nf * 16 + llo;
        float e = __expf(2.0f * a[v]);                   // exp(sim / tau), tau=0.5
        if (diagOut && grow == gcol) diagOut[grow] = e;
        if (posM) {
          size_t ij = (size_t)grow * N + gcol;
          size_t ji = (size_t)gcol * N + grow;
          bool offd = (grow != gcol);
          if (posM[ij]) pP += e;
          if (offd && negM[ij]) pN += e;
          if (posM[ji]) pPT += e;
          if (offd && negM[ji]) pNT += e;
        }
        // sum e across the 16 lanes holding this row's columns
        float es = e;
#pragma unroll
        for (int d = 1; d < 16; d <<= 1) es += __shfl_xor(es, d, 32);
        if (llo == 0) atomicAdd(&rowAcc[rl], es);
      }
    }
  if (posM) {
#pragma unroll
    for (int d = 1; d < 32; d <<= 1) {
      pP  += __shfl_xor(pP,  d, 32);
      pN  += __shfl_xor(pN,  d, 32);
      pPT += __shfl_xor(pPT, d, 32);
      pNT += __shfl_xor(pNT, d, 32);
    }
    if (lane == 0) {
      atomicAdd(&mAcc[0], pP);  atomicAdd(&mAcc[1], pN);
      atomicAdd(&mAcc[2], pPT); atomicAdd(&mAcc[3], pNT);
    }
  }
  __syncthreads();
  if (threadIdx.x < 128)
    partialRow[(size_t)blockIdx.x * N + tileRow + threadIdx.x] = rowAcc[threadIdx.x];
  if (posM && threadIdx.x == 0) {
    int tile = blockIdx.y * gridDim.x + blockIdx.x;
    maskPartial[tile * 4 + 0] = mAcc[0];
    maskPartial[tile * 4 + 1] = mAcc[1];
    maskPartial[tile * 4 + 2] = mAcc[2];
    maskPartial[tile * 4 + 3] = mAcc[3];
  }
}

__global__ void reduce_cols(const float* __restrict__ partial, float* __restrict__ out,
                            int N, int parts) {
  int i = blockIdx.x * blockDim.x + threadIdx.x;
  if (i < N) {
    float s = 0.f;
    for (int p = 0; p < parts; ++p) s += partial[(size_t)p * N + i];  // fixed order
    out[i] = s;
  }
}

__global__ __launch_bounds__(256) void finalize(const float* __restrict__ diagB,
                                                const float* __restrict__ R1,
                                                const float* __restrict__ R2,
                                                const float* __restrict__ Bs,
                                                const float* __restrict__ BTs,
                                                const float* __restrict__ dR1,
                                                const float* __restrict__ dR2,
                                                const float* __restrict__ maskPartial,
                                                int tiles, int N, float* __restrict__ out) {
  __shared__ float redA[256];
  __shared__ float redM[4][256];
  const int tid = threadIdx.x;
  float s = 0.f;
  for (int i = tid; i < N; i += 256) {
    float num = diagB[i];  // exp(2*S12[i,i])
    float l1 = logf(R1[i] + Bs[i]  - dR1[i]) - logf(num);
    float l2 = logf(R2[i] + BTs[i] - dR2[i]) - logf(num);
    s += 0.5f * (l1 + l2);
  }
  float m0 = 0.f, m1 = 0.f, m2 = 0.f, m3 = 0.f;
  for (int t = tid; t < tiles; t += 256) {
    m0 += maskPartial[t * 4 + 0];
    m1 += maskPartial[t * 4 + 1];
    m2 += maskPartial[t * 4 + 2];
    m3 += maskPartial[t * 4 + 3];
  }
  redA[tid] = s;
  redM[0][tid] = m0; redM[1][tid] = m1; redM[2][tid] = m2; redM[3][tid] = m3;
  __syncthreads();
  for (int st = 128; st > 0; st >>= 1) {
    if (tid < st) {
      redA[tid] += redA[tid + st];
      redM[0][tid] += redM[0][tid + st];
      redM[1][tid] += redM[1][tid + st];
      redM[2][tid] += redM[2][tid + st];
      redM[3][tid] += redM[3][tid + st];
    }
    __syncthreads();
  }
  if (tid == 0) {
    float unsup = redA[0] / (float)N;
    float pos12 = redM[0][0], neg12 = redM[1][0];
    float pos21 = redM[2][0], neg21 = redM[3][0];
    float s1 = logf(pos12 + neg12) - logf(pos12);
    float s2 = logf(pos21 + neg21) - logf(pos21);
    out[0] = unsup;
    out[1] = 0.5f * (s1 + s2);
  }
}

extern "C" void kernel_launch(void* const* d_in, const int* in_sizes, int n_in,
                              void* d_out, int out_size, void* d_ws, size_t ws_size,
                              hipStream_t stream) {
  (void)in_sizes; (void)n_in; (void)out_size; (void)ws_size;
  const int N = N_ROWS, D = D_DIM;
  const float* z1 = (const float*)d_in[0];
  const float* z2 = (const float*)d_in[1];
  const unsigned char* posM = (const unsigned char*)d_in[2];  // numpy bool layout
  const unsigned char* negM = (const unsigned char*)d_in[3];
  const float* W1 = (const float*)d_in[4];
  const float* b1 = (const float*)d_in[5];
  const float* W2 = (const float*)d_in[6];
  const float* b2 = (const float*)d_in[7];
  float* out = (float*)d_out;

  char* base = (char*)d_ws;
  size_t off = 0;
  auto alloc = [&](size_t bytes) -> void* {
    void* p = base + off;
    off += (bytes + 255) & ~(size_t)255;
    return p;
  };
  // persistent across whole pipeline (~16.6 MB)
  _Float16* n1h = (_Float16*)alloc((size_t)N * D * 2);
  _Float16* n2h = (_Float16*)alloc((size_t)N * D * 2);
  float* diagR1 = (float*)alloc((size_t)N * 4);
  float* diagR2 = (float*)alloc((size_t)N * 4);
  float* diagB  = (float*)alloc((size_t)N * 4);
  float* R1  = (float*)alloc((size_t)N * 4);
  float* R2  = (float*)alloc((size_t)N * 4);
  float* Bs  = (float*)alloc((size_t)N * 4);
  float* BTs = (float*)alloc((size_t)N * 4);
  const int TILES = (N / 128) * (N / 128);
  float* maskPartial = (float*)alloc((size_t)TILES * 4 * 4);
  // projection-phase scratch (~33 MB); partialRow aliases zh after projections
  _Float16* w1h = (_Float16*)alloc((size_t)D * D * 2);
  _Float16* w2h = (_Float16*)alloc((size_t)D * D * 2);
  _Float16* zh  = (_Float16*)alloc((size_t)N * D * 2);
  _Float16* th  = (_Float16*)alloc((size_t)N * D * 2);
  float*    hf  = (float*)alloc((size_t)N * D * 4);
  float* partialRow = (float*)zh;  // 64*N*4 = 2 MB <= 8 MB, reused post-projection

  dim3 blk(256);
  f32_to_h<<<dim3((D * D + 255) / 256), blk, 0, stream>>>(W1, w1h, D * D);
  f32_to_h<<<dim3((D * D + 255) / 256), blk, 0, stream>>>(W2, w2h, D * D);

  dim3 ggrid(D / 128, N / 128);
  for (int s = 0; s < 2; ++s) {
    const float* z = s ? z2 : z1;
    _Float16* nh = s ? n2h : n1h;
    f32_to_h<<<dim3((N * D + 255) / 256), blk, 0, stream>>>(z, zh, N * D);
    gemm_nt_bias<0><<<ggrid, blk, 0, stream>>>(zh, w1h, b1, th, nullptr, N, D, D);
    gemm_nt_bias<1><<<ggrid, blk, 0, stream>>>(th, w2h, b2, nullptr, hf, N, D, D);
    normalize_rows<<<dim3(N), blk, 0, stream>>>(hf, nh, D);
  }

  dim3 sgrid(N / 128, N / 128);
  // S11: row sums + diag
  sim_pass<<<sgrid, blk, 0, stream>>>(n1h, n1h, partialRow, diagR1, nullptr, nullptr, nullptr, N, D);
  reduce_cols<<<dim3(N / 256), blk, 0, stream>>>(partialRow, R1, N, N / 128);
  // S22: row sums + diag
  sim_pass<<<sgrid, blk, 0, stream>>>(n2h, n2h, partialRow, diagR2, nullptr, nullptr, nullptr, N, D);
  reduce_cols<<<dim3(N / 256), blk, 0, stream>>>(partialRow, R2, N, N / 128);
  // S12: row sums + diag + both mask orientations
  sim_pass<<<sgrid, blk, 0, stream>>>(n1h, n2h, partialRow, diagB, posM, negM, maskPartial, N, D);
  reduce_cols<<<dim3(N / 256), blk, 0, stream>>>(partialRow, Bs, N, N / 128);
  // S21 = S12^T: row sums give column sums of S12
  sim_pass<<<sgrid, blk, 0, stream>>>(n2h, n1h, partialRow, nullptr, nullptr, nullptr, nullptr, N, D);
  reduce_cols<<<dim3(N / 256), blk, 0, stream>>>(partialRow, BTs, N, N / 128);

  finalize<<<dim3(1), blk, 0, stream>>>(diagB, R1, R2, Bs, BTs, diagR1, diagR2,
                                        maskPartial, TILES, N, out);
}
